// NonLocalBlock_4337916969538
// MI455X (gfx1250) — matrix operand order repaired and checked by reading, not compiled, not run on hardware
//
#include <hip/hip_runtime.h>

#define BB 8
#define CC 256
#define CID 128
#define NNN 4096

typedef __attribute__((ext_vector_type(16))) __bf16 v16bf;
typedef __attribute__((ext_vector_type(8)))  __bf16 v8bf;
typedef __attribute__((ext_vector_type(8)))  float  v8f;

union Frag16 { v16bf v; v8bf h[2]; __bf16 e[16]; };
union Pack8  { v8bf v; __bf16 e[8]; };

#if defined(__AMDGCN__)
#define USE_ASYNC_LDS 1
#endif

// Copy 16B global -> LDS via GLOBAL_LOAD_ASYNC_TO_LDS_B128 (ASYNCcnt-tracked,
// no VGPR round trip). VDST operand is the LDS byte address; generic LDS
// pointers carry the LDS offset in the low 32 bits (ISA aperture mapping).
__device__ __forceinline__ void copy16_g2l(__bf16* dst_lds, const __bf16* src_g) {
#ifdef USE_ASYNC_LDS
  unsigned lds = (unsigned)(unsigned long long)dst_lds;
  unsigned long long ga = (unsigned long long)src_g;
  asm volatile("global_load_async_to_lds_b128 %0, %1, off"
               :
               : "v"(lds), "v"(ga)
               : "memory");
#else
  *(uint4*)dst_lds = *(const uint4*)src_g;
#endif
}

__device__ __forceinline__ void wait_async_copies() {
#ifdef USE_ASYNC_LDS
#if __has_builtin(__builtin_amdgcn_s_wait_asynccnt)
  __builtin_amdgcn_s_wait_asynccnt(0);
#else
  asm volatile("s_wait_asynccnt 0x0" ::: "memory");
#endif
#endif
}

__device__ __forceinline__ v8f wmma_bf16(const Frag16& a, const Frag16& b, v8f c) {
  return __builtin_amdgcn_wmma_f32_16x16x32_bf16(false, a.v, false, b.v, (short)0, c,
                                                 false, false);
}

__device__ __forceinline__ float rmax16(float v) {
  v = fmaxf(v, __shfl_xor(v, 1));
  v = fmaxf(v, __shfl_xor(v, 2));
  v = fmaxf(v, __shfl_xor(v, 4));
  v = fmaxf(v, __shfl_xor(v, 8));
  return v;
}
__device__ __forceinline__ float rsum16(float v) {
  v += __shfl_xor(v, 1);
  v += __shfl_xor(v, 2);
  v += __shfl_xor(v, 4);
  v += __shfl_xor(v, 8);
  return v;
}

// ---------------- conversion kernels ----------------

// Tiled transpose: xT[b][n][c] = (bf16) x[b][c][n].  64x64 tile via LDS so both
// the f32 reads and bf16 writes are fully coalesced.
__global__ __launch_bounds__(256) void k_transpose_x(const float* __restrict__ x,
                                                     __bf16* __restrict__ xT) {
  __shared__ __bf16 tile[64][65];  // pad to avoid bank conflicts on transposed read
  const int b = blockIdx.z;
  const int c0 = blockIdx.y * 64;
  const int n0 = blockIdx.x * 64;
#pragma unroll
  for (int t = threadIdx.x; t < 4096; t += 256) {
    int cc = t >> 6, nn = t & 63;
    tile[cc][nn] = (__bf16)x[((size_t)b * CC + c0 + cc) * NNN + n0 + nn];
  }
  __syncthreads();
#pragma unroll
  for (int t = threadIdx.x; t < 4096; t += 256) {
    int nn = t >> 6, cc = t & 63;
    xT[((size_t)b * NNN + n0 + nn) * CC + c0 + cc] = tile[cc][nn];
  }
}

__global__ __launch_bounds__(256) void k_cvt_bf16(const float* __restrict__ s,
                                                  __bf16* __restrict__ d, int n) {
  int i = blockIdx.x * 256 + threadIdx.x;
  if (i < n) d[i] = (__bf16)s[i];
}

// ---------------- projection GEMM: theta/phi/g ----------------
// out(proj)[n][ci] = sum_c xT[n][c] * w[ci][c] + bias[ci]
// theta/phi stored transposed [N][CI]; g stored [CI][N].
__global__ __launch_bounds__(256) void k_proj(const __bf16* __restrict__ xT,
                                              const __bf16* __restrict__ wAll, // [3][CI][C]: g,theta,phi
                                              const float* __restrict__ g_b,
                                              const float* __restrict__ theta_b,
                                              const float* __restrict__ phi_b,
                                              __bf16* __restrict__ thetaT,
                                              __bf16* __restrict__ phiT,
                                              __bf16* __restrict__ gmat) {
  const int which = blockIdx.z;  // 0=g, 1=theta, 2=phi
  const int b = blockIdx.y;
  const int wid = threadIdx.x >> 5;
  const int lane = threadIdx.x & 31;
  const int half = lane >> 4;
  const int n_c = lane & 15;

  const int tile = blockIdx.x * 8 + wid;  // (N/16)*(CI/16) = 256*8 tiles
  const int i0 = (tile >> 3) * 16;        // row block in N
  const int ci0 = (tile & 7) * 16;        // col block in CI

  const __bf16* w = wAll + (size_t)which * CID * CC;
  const __bf16* xrow = xT + ((size_t)b * NNN + i0 + n_c) * CC;
  const __bf16* wrow = w + (size_t)(ci0 + n_c) * CC;

  v8f acc = {0.f, 0.f, 0.f, 0.f, 0.f, 0.f, 0.f, 0.f};
#pragma unroll
  for (int kb = 0; kb < CC; kb += 32) {
    Frag16 a, bb;
    a.h[0] = *(const v8bf*)(xrow + kb + half * 16);
    a.h[1] = *(const v8bf*)(xrow + kb + half * 16 + 8);
    bb.h[0] = *(const v8bf*)(wrow + kb + half * 16);
    bb.h[1] = *(const v8bf*)(wrow + kb + half * 16 + 8);
    acc = wmma_bf16(a, bb, acc);
  }

  const float* bias = (which == 0) ? g_b : (which == 1 ? theta_b : phi_b);
  const float bv = bias[ci0 + n_c];

  if (which == 0) {  // g: [CI][N] -- each lane's 8 rows are contiguous: one b128 store
    Pack8 p;
#pragma unroll
    for (int r = 0; r < 8; ++r) p.e[r] = (__bf16)(acc[r] + bv);
    *(v8bf*)(gmat + ((size_t)b * CID + ci0 + n_c) * NNN + i0 + 8 * half) = p.v;
  } else {
    __bf16* dst = (which == 1) ? thetaT : phiT;
#pragma unroll
    for (int r = 0; r < 8; ++r) {
      int row = i0 + r + 8 * half;
      dst[((size_t)b * NNN + row) * CID + ci0 + n_c] = (__bf16)(acc[r] + bv);
    }
  }
}

// ---------------- fused flash attention ----------------
// yT[n][ci] = (1/l_n) * sum_j exp(S[n][j]-m_n) * g[ci][j],  S = thetaT * phiT^T
__global__ __launch_bounds__(256) void k_attn(const __bf16* __restrict__ thetaT,
                                              const __bf16* __restrict__ phiT,
                                              const __bf16* __restrict__ gmat,
                                              __bf16* __restrict__ yT) {
  const int b = blockIdx.y;
  const int wid = threadIdx.x >> 5;
  const int lane = threadIdx.x & 31;
  const int half = lane >> 4;
  const int n_c = lane & 15;
  const int i0 = blockIdx.x * 128 + wid * 16;  // this wave's 16 query rows

  __shared__ __bf16 phi_s[32][CID];      // j-slab of phiT (rows j, K=ci contiguous)
  __shared__ __bf16 g_s[CID][32];        // j-slab of g (rows ci, K=j contiguous)
  __shared__ __bf16 P_s[8][16][40];      // per-wave P tile, 80B row stride (conflict-free)

  // Preload theta A-fragments for this wave's rows (K = CI = 128 -> 4 frags)
  Frag16 ta[4];
  {
    const __bf16* trow = thetaT + ((size_t)b * NNN + i0 + n_c) * CID;
#pragma unroll
    for (int kb = 0; kb < 4; ++kb) {
      ta[kb].h[0] = *(const v8bf*)(trow + kb * 32 + half * 16);
      ta[kb].h[1] = *(const v8bf*)(trow + kb * 32 + half * 16 + 8);
    }
  }

  v8f yacc[8];
#pragma unroll
  for (int t = 0; t < 8; ++t) yacc[t] = (v8f){0.f, 0.f, 0.f, 0.f, 0.f, 0.f, 0.f, 0.f};
  float mrow[8], lrow[8];
#pragma unroll
  for (int r = 0; r < 8; ++r) { mrow[r] = -3.0e38f; lrow[r] = 0.f; }

  for (int j0 = 0; j0 < NNN; j0 += 32) {
    __syncthreads();  // previous slab fully consumed by all waves
    // stage phiT[j0..j0+31][0..127] (contiguous 8KB) -- async to LDS
    {
      const __bf16* src = phiT + ((size_t)b * NNN + j0) * CID;
      __bf16* dst = &phi_s[0][0];
#pragma unroll
      for (int t = threadIdx.x; t < 512; t += 256)
        copy16_g2l(dst + t * 8, src + t * 8);
    }
    // stage g[0..127][j0..j0+31] -- async to LDS
    {
#pragma unroll
      for (int t = threadIdx.x; t < 512; t += 256) {
        int row = t >> 2, c16 = t & 3;
        copy16_g2l(&g_s[row][c16 * 8],
                   gmat + ((size_t)b * CID + row) * NNN + j0 + c16 * 8);
      }
    }
    wait_async_copies();
    __syncthreads();

    // S tiles: two 16x16 tiles over j (K = CI = 128)
    v8f s[2];
#pragma unroll
    for (int tj = 0; tj < 2; ++tj) {
      v8f acc = {0.f, 0.f, 0.f, 0.f, 0.f, 0.f, 0.f, 0.f};
#pragma unroll
      for (int kb = 0; kb < 4; ++kb) {
        Frag16 bf;
        const __bf16* prow = &phi_s[tj * 16 + n_c][kb * 32 + half * 16];
        bf.h[0] = *(const v8bf*)prow;
        bf.h[1] = *(const v8bf*)(prow + 8);
        acc = wmma_bf16(ta[kb], bf, acc);
      }
      s[tj] = acc;
    }

    // online softmax update (row r in slot r, row index = r + 8*half)
    float alpha[8];
#pragma unroll
    for (int r = 0; r < 8; ++r) {
      float t = rmax16(fmaxf(s[0][r], s[1][r]));
      float mn = fmaxf(mrow[r], t);
      alpha[r] = __expf(mrow[r] - mn);
      mrow[r] = mn;
      float p0 = __expf(s[0][r] - mn);
      float p1 = __expf(s[1][r] - mn);
      s[0][r] = p0;
      s[1][r] = p1;
      lrow[r] = lrow[r] * alpha[r] + rsum16(p0 + p1);
    }
#pragma unroll
    for (int t = 0; t < 8; ++t)
#pragma unroll
      for (int r = 0; r < 8; ++r) yacc[t][r] *= alpha[r];

    // spill P tile to LDS (D layout) and reload as A fragment
#pragma unroll
    for (int r = 0; r < 8; ++r) {
      P_s[wid][r + 8 * half][n_c] = (__bf16)s[0][r];
      P_s[wid][r + 8 * half][16 + n_c] = (__bf16)s[1][r];
    }
    Frag16 pa;
    pa.h[0] = *(const v8bf*)&P_s[wid][n_c][half * 16];
    pa.h[1] = *(const v8bf*)&P_s[wid][n_c][half * 16 + 8];

    // y += P * g_slab  (8 ci-tiles, K = 32)
#pragma unroll
    for (int tc = 0; tc < 8; ++tc) {
      Frag16 gb;
      const __bf16* grow = &g_s[tc * 16 + n_c][half * 16];
      gb.h[0] = *(const v8bf*)grow;
      gb.h[1] = *(const v8bf*)(grow + 8);
      yacc[tc] = wmma_bf16(pa, gb, yacc[tc]);
    }
  }

  // finalize: yT[row][ci] = yacc / l
#pragma unroll
  for (int r = 0; r < 8; ++r) {
    float inv = 1.0f / lrow[r];
    int row = i0 + r + 8 * half;
#pragma unroll
    for (int tc = 0; tc < 8; ++tc)
      yT[((size_t)b * NNN + row) * CID + tc * 16 + n_c] = (__bf16)(yacc[tc][r] * inv);
  }
}

// ---------------- output projection + residual ----------------
// out[b][c][n] = sum_ci W[c][ci] * yT[n][ci] + W_b[c] + x[b][c][n]
__global__ __launch_bounds__(256) void k_out(const __bf16* __restrict__ yT,
                                             const __bf16* __restrict__ Wbf, // [C][CI]
                                             const float* __restrict__ W_b,
                                             const float* __restrict__ x,
                                             float* __restrict__ out) {
  const int b = blockIdx.y;
  const int wid = threadIdx.x >> 5;
  const int lane = threadIdx.x & 31;
  const int half = lane >> 4;
  const int n_c = lane & 15;

  const int tile = blockIdx.x * 8 + wid;  // (N/16)*(C/16) = 256*16 tiles
  const int i0 = (tile >> 4) * 16;        // row block in N
  const int c0 = (tile & 15) * 16;        // col block in C

  const __bf16* yrow = yT + ((size_t)b * NNN + i0 + n_c) * CID;
  const __bf16* wrow = Wbf + (size_t)(c0 + n_c) * CID;

  v8f acc = {0.f, 0.f, 0.f, 0.f, 0.f, 0.f, 0.f, 0.f};
#pragma unroll
  for (int kb = 0; kb < CID; kb += 32) {
    Frag16 a, bb;
    a.h[0] = *(const v8bf*)(yrow + kb + half * 16);
    a.h[1] = *(const v8bf*)(yrow + kb + half * 16 + 8);
    bb.h[0] = *(const v8bf*)(wrow + kb + half * 16);
    bb.h[1] = *(const v8bf*)(wrow + kb + half * 16 + 8);
    acc = wmma_bf16(a, bb, acc);
  }

  // Each lane's 8 output rows are contiguous f32: packed 128-bit residual
  // loads + stores instead of 16 scattered scalar accesses.
  const float bv = W_b[c0 + n_c];
  size_t base = ((size_t)b * CC + c0 + n_c) * NNN + i0 + 8 * half;
  float4 x0 = *(const float4*)(x + base);
  float4 x1 = *(const float4*)(x + base + 4);
  float4 o0 = {acc[0] + bv + x0.x, acc[1] + bv + x0.y,
               acc[2] + bv + x0.z, acc[3] + bv + x0.w};
  float4 o1 = {acc[4] + bv + x1.x, acc[5] + bv + x1.y,
               acc[6] + bv + x1.z, acc[7] + bv + x1.w};
  *(float4*)(out + base) = o0;
  *(float4*)(out + base + 4) = o1;
}

// ---------------- host launcher ----------------
extern "C" void kernel_launch(void* const* d_in, const int* in_sizes, int n_in,
                              void* d_out, int out_size, void* d_ws, size_t ws_size,
                              hipStream_t stream) {
  const float* x       = (const float*)d_in[0];
  const float* g_w     = (const float*)d_in[1];
  const float* g_b     = (const float*)d_in[2];
  const float* theta_w = (const float*)d_in[3];
  const float* theta_b = (const float*)d_in[4];
  const float* phi_w   = (const float*)d_in[5];
  const float* phi_b   = (const float*)d_in[6];
  const float* W_w     = (const float*)d_in[7];
  const float* W_b     = (const float*)d_in[8];
  float* out = (float*)d_out;

  char* ws = (char*)d_ws;
  size_t off = 0;
  __bf16* xT = (__bf16*)(ws + off);      off += (size_t)BB * NNN * CC * 2;   // 16 MB
  __bf16* thetaT = (__bf16*)(ws + off);  off += (size_t)BB * NNN * CID * 2;  // 8 MB
  __bf16* phiT = (__bf16*)(ws + off);    off += (size_t)BB * NNN * CID * 2;
  __bf16* gmat = (__bf16*)(ws + off);    off += (size_t)BB * CID * NNN * 2;
  __bf16* yT = (__bf16*)(ws + off);      off += (size_t)BB * NNN * CID * 2;
  __bf16* wAll = (__bf16*)(ws + off);    off += (size_t)3 * CID * CC * 2;    // g,theta,phi
  __bf16* Wbf = (__bf16*)(ws + off);     off += (size_t)CC * CID * 2;

  // 1) convert / transpose (tiled, coalesced both sides)
  {
    k_transpose_x<<<dim3(NNN / 64, CC / 64, BB), dim3(256), 0, stream>>>(x, xT);
    int wn = CID * CC;
    k_cvt_bf16<<<dim3((wn + 255) / 256), dim3(256), 0, stream>>>(g_w, wAll, wn);
    k_cvt_bf16<<<dim3((wn + 255) / 256), dim3(256), 0, stream>>>(theta_w, wAll + wn, wn);
    k_cvt_bf16<<<dim3((wn + 255) / 256), dim3(256), 0, stream>>>(phi_w, wAll + 2 * wn, wn);
    int on = CC * CID;
    k_cvt_bf16<<<dim3((on + 255) / 256), dim3(256), 0, stream>>>(W_w, Wbf, on);
  }
  // 2) projections: grid = (2048 tiles / 8 waves, B, 3)
  k_proj<<<dim3(256, BB, 3), dim3(256), 0, stream>>>(xT, wAll, g_b, theta_b, phi_b,
                                                     thetaT, phiT, gmat);
  // 3) fused attention: each block handles 128 query rows
  k_attn<<<dim3(NNN / 128, BB), dim3(256), 0, stream>>>(thetaT, phiT, gmat, yT);
  // 4) output projection + residual
  k_out<<<dim3(512, BB), dim3(256), 0, stream>>>(yT, Wbf, W_b, x, out);

  (void)in_sizes; (void)n_in; (void)out_size; (void)ws_size; (void)off;
}
